// GraphSAGE_22170621182410
// MI455X (gfx1250) — compile-verified, matrix operand order checked by reading, not run on hardware
//
#include <hip/hip_runtime.h>
#include <stdint.h>

typedef __attribute__((ext_vector_type(16))) __bf16 v16bf;
typedef __attribute__((ext_vector_type(8)))  __bf16 v8bf;
typedef __attribute__((ext_vector_type(8)))  float  v8f;

// ---- bf16 <-> f32 helpers (bit-level) ----
__device__ __forceinline__ __bf16 f2bf(float f) {
  union { float f; uint32_t u; } cv; cv.f = f;
  uint32_t u = cv.u + 0x7fffu + ((cv.u >> 16) & 1u);   // round-to-nearest-even
  union { uint16_t s; __bf16 b; } out; out.s = (uint16_t)(u >> 16);
  return out.b;
}
__device__ __forceinline__ float bf2f(__bf16 b) {
  union { __bf16 b; uint16_t s; } in; in.b = b;
  union { uint32_t u; float f; } cv; cv.u = ((uint32_t)in.s) << 16;
  return cv.f;
}

// ---- degree / normalization ----
__global__ __launch_bounds__(256) void k_degree(const int* __restrict__ dst,
                                                int* __restrict__ deg, int E) {
  int i = blockIdx.x * 256 + threadIdx.x;
  if (i < E) atomicAdd(&deg[dst[i]], 1);
}

__global__ __launch_bounds__(256) void k_deginv(const int* __restrict__ deg,
                                                float* __restrict__ dinv, int N) {
  int i = blockIdx.x * 256 + threadIdx.x;
  if (i < N) dinv[i] = 1.0f / fmaxf((float)deg[i], 1.0f);
}

__global__ __launch_bounds__(256) void k_f32_to_bf16(const float* __restrict__ in,
                                                     __bf16* __restrict__ out, long long n) {
  long long i = (long long)blockIdx.x * 256 + threadIdx.x;
  if (i < n) out[i] = f2bf(in[i]);
}

// ---- edge scatter-add: agg[dst] += h[src]  (bf16 gather, f32 atomics -> L2) ----
__global__ __launch_bounds__(256) void k_aggregate(const __bf16* __restrict__ h,
                                                   const int* __restrict__ src,
                                                   const int* __restrict__ dst,
                                                   float* __restrict__ agg,
                                                   long long total, int log2din) {
  long long i = (long long)blockIdx.x * 256 + threadIdx.x;
  if (i >= total) return;
  int din = 1 << log2din;
  int e = (int)(i >> log2din);
  int c = (int)(i & (din - 1));
  float v = bf2f(h[((size_t)src[e] << log2din) + c]);
  unsafeAtomicAdd(&agg[((size_t)dst[e] << log2din) + c], v);
}

// ---- scale by deg_inv and convert to bf16 activation buffer ----
__global__ __launch_bounds__(256) void k_scale(const float* __restrict__ agg,
                                               const float* __restrict__ dinv,
                                               __bf16* __restrict__ aggBf,
                                               long long total, int log2din) {
  long long i = (long long)blockIdx.x * 256 + threadIdx.x;
  if (i >= total) return;
  aggBf[i] = f2bf(agg[i] * dinv[i >> log2din]);
}

// ---- pack [Wl ; Wr] into WMMA B-fragment order: [nt][kstep][lane][16 halves] ----
__global__ __launch_bounds__(256) void k_wpack(const float* __restrict__ Wl,
                                               const float* __restrict__ Wr,
                                               __bf16* __restrict__ Wpack,
                                               int din, int dout, int nT) {
  int i = blockIdx.x * 256 + threadIdx.x;
  int nK = (din << 1) >> 5;                 // K' / 32
  int total = nT * nK * 32 * 16;
  if (i >= total) return;
  int j    = i & 15;
  int lane = (i >> 4) & 31;
  int ksnt = i >> 9;
  int ks   = ksnt % nK;
  int nt   = ksnt / nK;
  int col  = (nt << 4) + (lane & 15);
  int kg   = (lane >> 4) << 3;
  int K    = (ks << 5) + ((j & 8) << 1) + kg + (j & 7);  // halves 0-7: +kg+j, 8-15: +16+kg+(j-8)
  float v = 0.0f;
  if (col < dout) v = (K < din) ? Wl[K * dout + col] : Wr[(K - din) * dout + col];
  Wpack[i] = f2bf(v);
}

// ---- fused GEMM: D = [agg_scaled | h] @ [Wl;Wr] + b, optional ReLU ----
// One wave32 handles 4 M-tiles x one N-tile. Per k-step: 10x global_load_b128
// (1 B pair + 4 A pairs) batched, then 4 independent WMMAs. Straight-line:
// no EXEC manipulation anywhere near the WMMAs.
#define MTILES 4
__global__ __launch_bounds__(256) void k_gemm(const __bf16* __restrict__ aggBf,
                                              const __bf16* __restrict__ h,
                                              const __bf16* __restrict__ Wpack,
                                              const float* __restrict__ bias,
                                              __bf16* __restrict__ outBf,
                                              float* __restrict__ outF,
                                              int M, int din, int dout, int relu) {
  int gtid = blockIdx.x * 256 + threadIdx.x;
  int wave = gtid >> 5;
  int lane = gtid & 31;
  int nT = (dout + 15) >> 4;
  int mT = M >> 4;                               // M multiple of 16
  int mG = (mT + MTILES - 1) / MTILES;
  if (wave >= mG * nT) return;                   // uniform within a wave
  int mg = wave / nT;
  int nt = wave - mg * nT;
  int r16 = lane & 15;
  int kg  = (lane >> 4) << 3;                    // 0 or 8
  int col = (nt << 4) + r16;
  bool colOK = col < dout;
  int half = din >> 5;                           // k-steps per source

  const __bf16* wp = Wpack + ((size_t)(nt * (half << 1)) * 32 + lane) * 16;
  const __bf16* arow[MTILES];
  const __bf16* hrow[MTILES];
#pragma unroll
  for (int t = 0; t < MTILES; ++t) {
    int mt = mg * MTILES + t;
    if (mt >= mT) mt = mT - 1;                   // clamp tail: redundant compute, store guarded
    int row = (mt << 4) + r16;
    arow[t] = aggBf + (size_t)row * din + kg;
    hrow[t] = h     + (size_t)row * din + kg;
  }

  v8f acc[MTILES];
#pragma unroll
  for (int t = 0; t < MTILES; ++t)
    acc[t] = (v8f){0.f, 0.f, 0.f, 0.f, 0.f, 0.f, 0.f, 0.f};

  // phase 1: A from scaled aggregate
  for (int ks = 0; ks < half; ++ks) {
    v8bf b0 = *(const v8bf*)(wp);
    v8bf b1 = *(const v8bf*)(wp + 8);
    wp += 512;
    v8bf a0[MTILES], a1[MTILES];
#pragma unroll
    for (int t = 0; t < MTILES; ++t) {
      a0[t] = *(const v8bf*)(arow[t]);
      a1[t] = *(const v8bf*)(arow[t] + 16);
      arow[t] += 32;
    }
    v16bf b = __builtin_shufflevector(b0, b1, 0,1,2,3,4,5,6,7,8,9,10,11,12,13,14,15);
#pragma unroll
    for (int t = 0; t < MTILES; ++t) {
      v16bf a = __builtin_shufflevector(a0[t], a1[t], 0,1,2,3,4,5,6,7,8,9,10,11,12,13,14,15);
      acc[t] = __builtin_amdgcn_wmma_f32_16x16x32_bf16(false, a, false, b,
                                                       (short)0, acc[t], false, false);
    }
  }
  // phase 2: A from previous-layer activations
  for (int ks = 0; ks < half; ++ks) {
    v8bf b0 = *(const v8bf*)(wp);
    v8bf b1 = *(const v8bf*)(wp + 8);
    wp += 512;
    v8bf a0[MTILES], a1[MTILES];
#pragma unroll
    for (int t = 0; t < MTILES; ++t) {
      a0[t] = *(const v8bf*)(hrow[t]);
      a1[t] = *(const v8bf*)(hrow[t] + 16);
      hrow[t] += 32;
    }
    v16bf b = __builtin_shufflevector(b0, b1, 0,1,2,3,4,5,6,7,8,9,10,11,12,13,14,15);
#pragma unroll
    for (int t = 0; t < MTILES; ++t) {
      v16bf a = __builtin_shufflevector(a0[t], a1[t], 0,1,2,3,4,5,6,7,8,9,10,11,12,13,14,15);
      acc[t] = __builtin_amdgcn_wmma_f32_16x16x32_bf16(false, a, false, b,
                                                       (short)0, acc[t], false, false);
    }
  }

  if (colOK) {
    float bv = bias[col];
    int hi8 = (lane >> 4) << 3;
#pragma unroll
    for (int t = 0; t < MTILES; ++t) {
      int mt = mg * MTILES + t;
      if (mt < mT) {
        int mBase = (mt << 4) + hi8;             // D layout: m = r + 8*(lane>>4)
#pragma unroll
        for (int r = 0; r < 8; ++r) {
          float v = acc[t][r] + bv;
          if (relu) v = fmaxf(v, 0.0f);
          size_t o = (size_t)(mBase + r) * dout + col;
          if (outF) outF[o] = v;
          else      outBf[o] = f2bf(v);
        }
      }
    }
  }
}

extern "C" void kernel_launch(void* const* d_in, const int* in_sizes, int n_in,
                              void* d_out, int out_size, void* d_ws, size_t ws_size,
                              hipStream_t stream) {
  (void)n_in; (void)out_size; (void)ws_size;

  const float* x    = (const float*)d_in[0];
  const int*   edge = (const int*)d_in[1];
  const int E = in_sizes[1] / 2;
  const int N = in_sizes[0] / 128;
  const int* src = edge;           // edge_index[0]
  const int* dst = edge + E;       // edge_index[1]

  const float* Wl[4]   = {(const float*)d_in[2],  (const float*)d_in[5],
                          (const float*)d_in[8],  (const float*)d_in[11]};
  const float* Wr[4]   = {(const float*)d_in[3],  (const float*)d_in[6],
                          (const float*)d_in[9],  (const float*)d_in[12]};
  const float* bias[4] = {(const float*)d_in[4],  (const float*)d_in[7],
                          (const float*)d_in[10], (const float*)d_in[13]};
  const int din[4]  = {128, 256, 128, 64};
  const int dlog[4] = {7, 8, 7, 6};
  const int dout[4] = {256, 128, 64, 2};

  char* ws = (char*)d_ws;
  size_t off = 0;
  auto carve = [&](size_t bytes) -> void* {
    void* p = ws + off;
    off += (bytes + 255) & ~(size_t)255;
    return p;
  };
  int*    deg   = (int*)   carve((size_t)N * 4);
  float*  dinv  = (float*) carve((size_t)N * 4);
  __bf16* h0    = (__bf16*)carve((size_t)N * 256 * 2);
  __bf16* h1    = (__bf16*)carve((size_t)N * 256 * 2);
  float*  agg   = (float*) carve((size_t)N * 256 * 4);   // f32 accumulation, L2-resident
  __bf16* aggBf = (__bf16*)carve((size_t)N * 256 * 2);
  __bf16* Wpack = (__bf16*)carve((size_t)512 * 1024);

  // degree + inverse
  (void)hipMemsetAsync(deg, 0, (size_t)N * 4, stream);
  k_degree<<<(E + 255) / 256, 256, 0, stream>>>(dst, deg, E);
  k_deginv<<<(N + 255) / 256, 256, 0, stream>>>(deg, dinv, N);

  // x -> bf16 activations
  long long xElems = (long long)N * 128;
  k_f32_to_bf16<<<(int)((xElems + 255) / 256), 256, 0, stream>>>(x, h0, xElems);

  const __bf16* hin = h0;
  for (int l = 0; l < 4; ++l) {
    __bf16* hout = (hin == h0) ? h1 : h0;
    size_t aggElems = (size_t)N * din[l];

    (void)hipMemsetAsync(agg, 0, aggElems * 4, stream);

    long long tot = (long long)E * din[l];
    k_aggregate<<<(int)((tot + 255) / 256), 256, 0, stream>>>(hin, src, dst, agg,
                                                              tot, dlog[l]);
    k_scale<<<(int)((aggElems + 255) / 256), 256, 0, stream>>>(agg, dinv, aggBf,
                                                               (long long)aggElems, dlog[l]);
    int nT = (dout[l] + 15) / 16;
    int nK = (2 * din[l]) / 32;
    int wTot = nT * nK * 32 * 16;
    k_wpack<<<(wTot + 255) / 256, 256, 0, stream>>>(Wl[l], Wr[l], Wpack,
                                                    din[l], dout[l], nT);

    int mT = N / 16;
    int mG = (mT + MTILES - 1) / MTILES;
    int waves = mG * nT;
    bool last = (l == 3);
    k_gemm<<<(waves + 7) / 8, 256, 0, stream>>>(aggBf, hin, Wpack, bias[l],
                                                last ? nullptr : hout,
                                                last ? (float*)d_out : nullptr,
                                                N, din[l], dout[l], last ? 0 : 1);
    hin = hout;
  }
}
